// TransformerBlock_82154134438216
// MI455X (gfx1250) — matrix operand order repaired and checked by reading, not compiled, not run on hardware
//
#include <hip/hip_runtime.h>
#include <hip/hip_bf16.h>
#include <math.h>

typedef __bf16 bf16;
typedef __attribute__((ext_vector_type(16))) __bf16 v16bf;
typedef __attribute__((ext_vector_type(8)))  float  v8f;

union V8F   { v8f  v; float e[8];  };
union V16B  { v16bf v; bf16 e[16]; uint4 q[2]; };

#define DEVINL __device__ __forceinline__

// ---------- bf16 helpers (round-to-nearest-even) ----------
DEVINL unsigned f2bfu(float f) {
    unsigned u = __builtin_bit_cast(unsigned, f);
    unsigned r = u + 0x7FFFu + ((u >> 16) & 1u);
    return r >> 16;
}
DEVINL bf16 f2bf(float f) {
    unsigned short h = (unsigned short)f2bfu(f);
    return __builtin_bit_cast(bf16, h);
}

// ---------- gfx1250 async global->LDS (ASYNCcnt) ----------
DEVINL unsigned lds_off(const void* p) { return (unsigned)(uintptr_t)p; }  // LDS offset: low 32 bits of the address
DEVINL void async_ld_b128(unsigned ldsaddr, const void* gaddr) {
    asm volatile("global_load_async_to_lds_b128 %0, %1, off"
                 :: "v"(ldsaddr), "v"(gaddr) : "memory");
}
DEVINL void wait_async0() { asm volatile("s_wait_asynccnt 0" ::: "memory"); }

// ---------- WMMA fragment loaders (16x16x32 bf16; contiguous k = 16*(lane/16) + e, same for A and B) ----------
// A 16x32 (MxK): lane<16 -> M=lane, K=e; lane>=16 -> K=16+e
// The run is contiguous -> two 16B vector loads.
DEVINL v16bf load_a_frag(const bf16* p, int ld) {
    const int lane = threadIdx.x & 31;
    const bf16* row = p + (size_t)(lane & 15) * ld + (lane >> 4) * 16;
    V16B a;
    a.q[0] = *(const uint4*)(row);
    a.q[1] = *(const uint4*)(row + 8);
    return a.v;
}
// B 32x16 (KxN) from row-major W[N,K] (K contiguous): lane<16 -> N=lane, K=e; lane>=16 -> K=16+e
DEVINL v16bf load_b_frag_nt(const bf16* p, int ld) {
    const int lane = threadIdx.x & 31;
    const bf16* src = p + (size_t)(lane & 15) * ld + (lane >> 4) * 16;
    V16B b;
    b.q[0] = *(const uint4*)(src);
    b.q[1] = *(const uint4*)(src + 8);
    return b.v;
}
// B 32x16 (KxN) from row-major V[K,N] in LDS (N contiguous, K strided)
DEVINL v16bf load_b_frag_nn(const bf16* p, int ld) {
    const int lane = threadIdx.x & 31;
    const bf16* src = p + (lane & 15) + (size_t)((lane >> 4) * 16) * ld;
    V16B b;
#pragma unroll
    for (int e = 0; e < 16; ++e) b.e[e] = src[(size_t)e * ld];
    return b.v;
}

DEVINL v8f wmma_bf16(v16bf a, v16bf b, v8f c) {
    return __builtin_amdgcn_wmma_f32_16x16x32_bf16(false, a, false, b, (short)0, c, false, false);
}

// ---------- f32 -> bf16 convert ----------
__global__ __launch_bounds__(256) void cvt_bf16_kernel(const float* __restrict__ in,
                                                       bf16* __restrict__ out, size_t n) {
    for (size_t i = 4 * ((size_t)blockIdx.x * blockDim.x + threadIdx.x); i < n;
         i += 4 * (size_t)gridDim.x * blockDim.x) {
        float4 v = *(const float4*)(in + i);
        uint2 o;
        o.x = f2bfu(v.x) | (f2bfu(v.y) << 16);
        o.y = f2bfu(v.z) | (f2bfu(v.w) << 16);
        *(uint2*)(out + i) = o;
    }
}

// ---------- LayerNorm (C=1024), one block per row, bf16 output ----------
__global__ __launch_bounds__(256) void ln_bf16_kernel(const float* __restrict__ x,
                                                      const float* __restrict__ g,
                                                      const float* __restrict__ bta,
                                                      bf16* __restrict__ out) {
    const int row = blockIdx.x;
    const int tid = threadIdx.x, lane = tid & 31, wave = tid >> 5;
    const float* xr = x + (size_t)row * 1024;
    const int base = tid * 4;
    float v4[4];
    float s = 0.f;
#pragma unroll
    for (int i = 0; i < 4; ++i) { v4[i] = xr[base + i]; s += v4[i]; }
    __shared__ float red[8];
#pragma unroll
    for (int off = 16; off >= 1; off >>= 1) s += __shfl_xor(s, off, 32);
    if (lane == 0) red[wave] = s;
    __syncthreads();
    float tot = 0.f;
#pragma unroll
    for (int i = 0; i < 8; ++i) tot += red[i];
    const float mean = tot * (1.f / 1024.f);
    __syncthreads();
    float vs = 0.f;
#pragma unroll
    for (int i = 0; i < 4; ++i) { float d = v4[i] - mean; vs += d * d; }
#pragma unroll
    for (int off = 16; off >= 1; off >>= 1) vs += __shfl_xor(vs, off, 32);
    if (lane == 0) red[wave] = vs;
    __syncthreads();
    float vtot = 0.f;
#pragma unroll
    for (int i = 0; i < 8; ++i) vtot += red[i];
    const float rstd = rsqrtf(vtot * (1.f / 1024.f) + 1e-5f);
    float o4[4];
#pragma unroll
    for (int i = 0; i < 4; ++i) o4[i] = (v4[i] - mean) * rstd * g[base + i] + bta[base + i];
    uint2 o;
    o.x = f2bfu(o4[0]) | (f2bfu(o4[1]) << 16);
    o.y = f2bfu(o4[2]) | (f2bfu(o4[3]) << 16);
    *(uint2*)(out + (size_t)row * 1024 + base) = o;
}

// ---------- Tiled WMMA GEMM: out[n,m] = act(A[n,:]·W[m,:] + bias[m]) (+res) ----------
// Block tile 64(N) x 256(M), 8 waves (2x4), wave tile 32x64 (2x4 fragments).
// K-step 32, double-buffered LDS filled by async global->LDS copies.
template <int GELU, int RES, int OUTB>
__global__ __launch_bounds__(256) void gemm_wmma_kernel(
    const bf16* __restrict__ A, const bf16* __restrict__ W, const float* __restrict__ bias,
    const float* __restrict__ res, float* __restrict__ outf, bf16* __restrict__ outb,
    int N, int M, int K) {
    const int tid = threadIdx.x;
    const int wave = tid >> 5, lane = tid & 31;
    const int wr = wave >> 2, wc = wave & 3;      // 2x4 wave grid
    const int rowBase = blockIdx.y * 64;
    const int colBase = blockIdx.x * 256;

    __shared__ bf16 As[2][64][40];
    __shared__ bf16 Ws[2][256][40];

    // per-thread async-copy chunk addressing (16B chunks)
    const int ar = tid >> 2, ac = (tid & 3) * 8;                 // A: 64x32 = 256 chunks
    const bf16* agp = A + (size_t)(rowBase + ar) * K + ac;
    int wrr[4], wcc[4];
    const bf16* wgp[4];
#pragma unroll
    for (int t = 0; t < 4; ++t) {                                // W: 256x32 = 1024 chunks
        int ch = tid + t * 256;
        wrr[t] = ch >> 2; wcc[t] = (ch & 3) * 8;
        wgp[t] = W + (size_t)(colBase + wrr[t]) * K + wcc[t];
    }

    V8F acc[2][4];
#pragma unroll
    for (int i = 0; i < 2; ++i)
#pragma unroll
        for (int j = 0; j < 4; ++j)
#pragma unroll
            for (int r = 0; r < 8; ++r) acc[i][j].e[r] = 0.f;

    // prologue: fill buffer 0
    async_ld_b128(lds_off(&As[0][ar][ac]), agp);
#pragma unroll
    for (int t = 0; t < 4; ++t)
        async_ld_b128(lds_off(&Ws[0][wrr[t]][wcc[t]]), wgp[t]);
    wait_async0();
    __syncthreads();

    for (int k0 = 0; k0 < K; k0 += 32) {
        const int buf = (k0 >> 5) & 1;
        if (k0 + 32 < K) {      // prefetch next K-tile into the other buffer
            async_ld_b128(lds_off(&As[buf ^ 1][ar][ac]), agp + k0 + 32);
#pragma unroll
            for (int t = 0; t < 4; ++t)
                async_ld_b128(lds_off(&Ws[buf ^ 1][wrr[t]][wcc[t]]), wgp[t] + k0 + 32);
        }
        v16bf af[2], bfr[4];
        af[0] = load_a_frag(&As[buf][wr * 32 + 0][0], 40);
        af[1] = load_a_frag(&As[buf][wr * 32 + 16][0], 40);
#pragma unroll
        for (int fc = 0; fc < 4; ++fc)
            bfr[fc] = load_b_frag_nt(&Ws[buf][wc * 64 + fc * 16][0], 40);
#pragma unroll
        for (int fr = 0; fr < 2; ++fr)
#pragma unroll
            for (int fc = 0; fc < 4; ++fc)
                acc[fr][fc].v = wmma_bf16(af[fr], bfr[fc], acc[fr][fc].v);
        wait_async0();
        __syncthreads();
    }

    const int half = lane >> 4, nn = lane & 15;
#pragma unroll
    for (int fr = 0; fr < 2; ++fr)
#pragma unroll
        for (int fc = 0; fc < 4; ++fc) {
#pragma unroll
            for (int r = 0; r < 8; ++r) {
                int row = rowBase + wr * 32 + fr * 16 + r + 8 * half;
                int col = colBase + wc * 64 + fc * 16 + nn;
                float v = acc[fr][fc].e[r] + bias[col];
                if (GELU) v = 0.5f * v * (1.0f + erff(v * 0.70710678118654752f));
                size_t idx = (size_t)row * M + col;
                if (RES) v += res[idx];
                if (OUTB) outb[idx] = f2bf(v);
                else      outf[idx] = v;
            }
        }
}

// ---------- Fused causal flash attention ----------
// qkv bf16 [B,T,3C]; one wave per 16-row q tile; H=16, D=64, T=2048, C=1024.
// V tiles staged via async global->LDS, overlapped with S/softmax.
__global__ __launch_bounds__(256) void attn_kernel(const bf16* __restrict__ qkv,
                                                   bf16* __restrict__ o) {
    const int T = 2048, C3 = 3072;
    const int wave = threadIdx.x >> 5, lane = threadIdx.x & 31;
    const int tile = blockIdx.x * 8 + wave;     // 0..8191
    const int qt = tile & 127;                  // T/16 tiles
    const int bh = tile >> 7;
    const int h = bh & 15, b = bh >> 4;
    const int q0 = qt * 16;
    const int half = lane >> 4, nn = lane & 15;

    __shared__ bf16 plds[8][16][40];     // per-wave P tile
    __shared__ bf16 vlds[8][32][72];     // per-wave V tile (32 keys x 64 d)
    bf16(*P)[40] = plds[wave];
    bf16(*Vs)[72] = vlds[wave];

    const bf16* qbase = qkv + ((size_t)(b * T + q0)) * C3 + h * 64;
    const bf16* kbase = qkv + (size_t)b * T * C3 + 1024 + h * 64;
    const bf16* vbase = qkv + (size_t)b * T * C3 + 2048 + h * 64;

    v16bf qf[2];
    qf[0] = load_a_frag(qbase + 0, C3);
    qf[1] = load_a_frag(qbase + 32, C3);

    float mrow[8], lrow[8];
    V8F Oacc[4];
#pragma unroll
    for (int r = 0; r < 8; ++r) { mrow[r] = -1e30f; lrow[r] = 0.f; }
#pragma unroll
    for (int g = 0; g < 4; ++g)
#pragma unroll
        for (int r = 0; r < 8; ++r) Oacc[g].e[r] = 0.f;

    for (int j0 = 0; j0 < q0 + 16; j0 += 32) {
        // stage V tile (32x64 bf16 = 256 x 16B chunks, 8 per lane) -> LDS, async
#pragma unroll
        for (int i = 0; i < 8; ++i) {
            int ch = lane + 32 * i;
            int vr = ch >> 3, vc = (ch & 7) * 8;
            async_ld_b128(lds_off(&Vs[vr][vc]), vbase + (size_t)(j0 + vr) * C3 + vc);
        }

        V8F S[2];
#pragma unroll
        for (int kk = 0; kk < 2; ++kk)
#pragma unroll
            for (int r = 0; r < 8; ++r) S[kk].e[r] = 0.f;
#pragma unroll
        for (int kk = 0; kk < 2; ++kk)
#pragma unroll
            for (int dd = 0; dd < 2; ++dd) {
                v16bf kf = load_b_frag_nt(kbase + (size_t)(j0 + kk * 16) * C3 + dd * 32, C3);
                S[kk].v = wmma_bf16(qf[dd], kf, S[kk].v);
            }
        // scale + causal mask
#pragma unroll
        for (int kk = 0; kk < 2; ++kk)
#pragma unroll
            for (int r = 0; r < 8; ++r) {
                float s = S[kk].e[r] * 0.125f;            // 1/sqrt(64)
                int col = j0 + kk * 16 + nn;
                int row = q0 + r + 8 * half;
                S[kk].e[r] = (col > row) ? -1e30f : s;
            }
        // online softmax per row (rows live across 16 lanes of each half-wave)
#pragma unroll
        for (int r = 0; r < 8; ++r) {
            float mx = fmaxf(S[0].e[r], S[1].e[r]);
#pragma unroll
            for (int off = 8; off >= 1; off >>= 1) mx = fmaxf(mx, __shfl_xor(mx, off, 32));
            float mnew = fmaxf(mrow[r], mx);
            float corr = __expf(mrow[r] - mnew);
            mrow[r] = mnew;
            float p0 = __expf(S[0].e[r] - mnew);
            float p1 = __expf(S[1].e[r] - mnew);
            float rs = p0 + p1;
#pragma unroll
            for (int off = 8; off >= 1; off >>= 1) rs += __shfl_xor(rs, off, 32);
            lrow[r] = lrow[r] * corr + rs;
#pragma unroll
            for (int g = 0; g < 4; ++g) Oacc[g].e[r] *= corr;
            P[r + 8 * half][nn] = f2bf(p0);
            P[r + 8 * half][16 + nn] = f2bf(p1);
        }
        // re-fragment P (per-wave LDS tile; DS ops in-order within a wave)
        v16bf pf = load_a_frag(&P[0][0], 40);
        wait_async0();   // V tile resident in LDS
#pragma unroll
        for (int g = 0; g < 4; ++g) {
            v16bf vf = load_b_frag_nn(&Vs[0][g * 16], 72);
            Oacc[g].v = wmma_bf16(pf, vf, Oacc[g].v);
        }
    }

    float recip[8];
#pragma unroll
    for (int r = 0; r < 8; ++r) recip[r] = 1.f / lrow[r];
#pragma unroll
    for (int g = 0; g < 4; ++g)
#pragma unroll
        for (int r = 0; r < 8; ++r) {
            int row = q0 + r + 8 * half;
            o[((size_t)(b * T + row)) * 1024 + h * 64 + g * 16 + nn] =
                f2bf(Oacc[g].e[r] * recip[r]);
        }
}

// ---------- launch ----------
extern "C" void kernel_launch(void* const* d_in, const int* in_sizes, int n_in,
                              void* d_out, int out_size, void* d_ws, size_t ws_size,
                              hipStream_t stream) {
    (void)in_sizes; (void)n_in; (void)out_size; (void)ws_size;
    const int C = 1024, FF = 4096, C3 = 3072;
    const size_t NT = 8192;   // B*T

    const float* x      = (const float*)d_in[0];
    const float* ln1_g  = (const float*)d_in[1];
    const float* ln1_b  = (const float*)d_in[2];
    const float* qkv_w  = (const float*)d_in[3];
    const float* qkv_b  = (const float*)d_in[4];
    const float* out_w  = (const float*)d_in[5];
    const float* out_b  = (const float*)d_in[6];
    const float* ln2_g  = (const float*)d_in[7];
    const float* ln2_b  = (const float*)d_in[8];
    const float* fc1_w  = (const float*)d_in[9];
    const float* fc1_b  = (const float*)d_in[10];
    const float* fc2_w  = (const float*)d_in[11];
    const float* fc2_b  = (const float*)d_in[12];
    float* out = (float*)d_out;

    char* w = (char*)d_ws;
    auto alloc = [&](size_t bytes) { void* p = (void*)w; w += (bytes + 255) & ~(size_t)255; return p; };
    bf16* wq  = (bf16*)alloc((size_t)C3 * C * 2);
    bf16* wo  = (bf16*)alloc((size_t)C  * C * 2);
    bf16* w1  = (bf16*)alloc((size_t)FF * C * 2);
    bf16* w2  = (bf16*)alloc((size_t)C  * FF * 2);
    bf16* h1  = (bf16*)alloc(NT * C * 2);
    bf16* qkv = (bf16*)alloc(NT * C3 * 2);
    bf16* ob  = (bf16*)alloc(NT * C * 2);
    float* x1 = (float*)alloc(NT * C * 4);
    bf16* h2  = (bf16*)alloc(NT * C * 2);
    bf16* h3  = (bf16*)alloc(NT * FF * 2);

    cvt_bf16_kernel<<<2048, 256, 0, stream>>>(qkv_w, wq, (size_t)C3 * C);
    cvt_bf16_kernel<<<2048, 256, 0, stream>>>(out_w, wo, (size_t)C * C);
    cvt_bf16_kernel<<<2048, 256, 0, stream>>>(fc1_w, w1, (size_t)FF * C);
    cvt_bf16_kernel<<<2048, 256, 0, stream>>>(fc2_w, w2, (size_t)C * FF);

    // --- attention sublayer ---
    ln_bf16_kernel<<<NT, 256, 0, stream>>>(x, ln1_g, ln1_b, h1);
    gemm_wmma_kernel<0, 0, 1><<<dim3(C3 / 256, NT / 64), 256, 0, stream>>>(
        h1, wq, qkv_b, nullptr, nullptr, qkv, (int)NT, C3, C);
    attn_kernel<<<1024, 256, 0, stream>>>(qkv, ob);
    gemm_wmma_kernel<0, 1, 0><<<dim3(C / 256, NT / 64), 256, 0, stream>>>(
        ob, wo, out_b, x, x1, nullptr, (int)NT, C, C);

    // --- MLP sublayer ---
    ln_bf16_kernel<<<NT, 256, 0, stream>>>(x1, ln2_g, ln2_b, h2);
    gemm_wmma_kernel<1, 0, 1><<<dim3(FF / 256, NT / 64), 256, 0, stream>>>(
        h2, w1, fc1_b, nullptr, nullptr, h3, (int)NT, FF, C);
    gemm_wmma_kernel<0, 1, 0><<<dim3(C / 256, NT / 64), 256, 0, stream>>>(
        h3, w2, fc2_b, x1, out, nullptr, (int)NT, C, FF);
}
